// GML_49881750175841
// MI455X (gfx1250) — matrix operand order repaired and checked by reading, not compiled, not run on hardware
//
#include <hip/hip_runtime.h>
#include <hip/hip_bf16.h>
#include <math.h>

// ---------------- problem constants (from reference) ----------------
constexpr int Bn    = 16;    // batch
constexpr int Nreal = 512;   // real points per side
constexpr int Nh    = 384;   // padding points (N1 == M1)
constexpr int Npad  = 896;   // Nreal + Nh
constexpr int Dd    = 256;   // feature dim
constexpr float NEGW = -100000.0f;

typedef __attribute__((ext_vector_type(16))) _Float16 v16h;
typedef __attribute__((ext_vector_type(8)))  _Float16 v8h;
typedef __attribute__((ext_vector_type(4)))  _Float16 v4h;
typedef __attribute__((ext_vector_type(8)))  float    v8f;
typedef __attribute__((ext_vector_type(4)))  float    v4f;

__device__ __forceinline__ float sum16(float v) {  // reduce within groups of 16 lanes
  v += __shfl_xor(v, 1, 32);
  v += __shfl_xor(v, 2, 32);
  v += __shfl_xor(v, 4, 32);
  v += __shfl_xor(v, 8, 32);
  return v;
}

// online log-sum-exp accumulate
__device__ __forceinline__ void lse_acc(float& m, float& s, float v) {
  if (v <= m) s += __expf(v - m);
  else        { s = s * __expf(m - v) + 1.f; m = v; }
}

// ---------------- init: zero accumulators + outputs ----------------
__global__ void init_zero(float* rowsum, float* colsum, float* out) {
  int i = blockIdx.x * blockDim.x + threadIdx.x;
  if (i < Bn * Npad) { rowsum[i] = 0.f; colsum[i] = 0.f; }
  if (i < 2) out[i] = 0.f;
}

// ------------- prep: L2-normalize -> f16, plus raw f16 x3/x4 -------------
// one wave per (side, b, i) row; b128 loads, b64 f16 stores
__global__ __launch_bounds__(256) void prep_norm(
    const float* __restrict__ x1, const float* __restrict__ x2,
    const float* __restrict__ x3, const float* __restrict__ x4,
    _Float16* __restrict__ f1h, _Float16* __restrict__ f2h,
    _Float16* __restrict__ x3h, _Float16* __restrict__ x4h)
{
  int wid  = (blockIdx.x * blockDim.x + threadIdx.x) >> 5;
  int lane = threadIdx.x & 31;
  int side = wid / (Bn * Npad);
  int r    = wid % (Bn * Npad);
  int b = r / Npad, i = r % Npad;

  const float* src;
  if (side == 0) src = (i < Nreal) ? x1 + ((size_t)b * Nreal + i) * Dd
                                   : x3 + ((size_t)b * Nh + (i - Nreal)) * Dd;
  else           src = (i < Nreal) ? x2 + ((size_t)b * Nreal + i) * Dd
                                   : x4 + ((size_t)b * Nh + (i - Nreal)) * Dd;
  const v4f* srcv = (const v4f*)src;             // 64 float4 per row
  v4f a0 = srcv[lane];
  v4f a1 = srcv[lane + 32];
  float ss = a0.x*a0.x + a0.y*a0.y + a0.z*a0.z + a0.w*a0.w
           + a1.x*a1.x + a1.y*a1.y + a1.z*a1.z + a1.w*a1.w;
  ss += __shfl_xor(ss, 16, 32);
  ss = sum16(ss);
  float sc = 1.0f / fmaxf(sqrtf(ss), 1e-12f);

  v4h* dstv = (v4h*)((side == 0 ? f1h : f2h) + ((size_t)b * Npad + i) * Dd);
  v4h o0, o1;
#pragma unroll
  for (int e = 0; e < 4; ++e) { o0[e] = (_Float16)(a0[e] * sc); o1[e] = (_Float16)(a1[e] * sc); }
  dstv[lane] = o0; dstv[lane + 32] = o1;

  if (i >= Nreal) {
    v4h* rawv = (v4h*)((side == 0 ? x3h : x4h) + ((size_t)b * Nh + (i - Nreal)) * Dd);
    v4h r0, r1;
#pragma unroll
    for (int e = 0; e < 4; ++e) { r0[e] = (_Float16)a0[e]; r1[e] = (_Float16)a1[e]; }
    rawv[lane] = r0; rawv[lane + 32] = r1;
  }
}

// ------------- sim GEMM: S = exp(10 * A@B^T) + fused row/col sums -------------
// one wave per 16x16 tile, K=256 -> 8 x v_wmma_f32_16x16x32_f16
__global__ __launch_bounds__(256) void sim_gemm(
    const _Float16* __restrict__ f1h, const _Float16* __restrict__ f2h,
    float* __restrict__ S, float* __restrict__ rowsum, float* __restrict__ colsum)
{
  const int TPD = Npad / 16;  // 56 tiles per dim
  int wid  = (blockIdx.x * blockDim.x + threadIdx.x) >> 5;
  int lane = threadIdx.x & 31;
  int b    = wid / (TPD * TPD);
  int rem  = wid % (TPD * TPD);
  int ti = rem / TPD, tj = rem % TPD;
  int laneM = lane & 15, hi = lane >> 4;

  const _Float16* Ar = f1h + ((size_t)b * Npad + ti * 16 + laneM) * Dd;
  const _Float16* Br = f2h + ((size_t)b * Npad + tj * 16 + laneM) * Dd;

  v8f acc = {};
#pragma unroll
  for (int k = 0; k < Dd; k += 32) {
    // A 16x32 layout (ISA 7.12.2): lane<16 -> K {0..7,16..23}; lane>=16 -> K {8..15,24..31}
    v8h alo = *(const v8h*)(Ar + k + hi * 8);
    v8h ahi = *(const v8h*)(Ar + k + hi * 8 + 16);
    v16h a;
#pragma unroll
    for (int e = 0; e < 8; ++e) { a[e] = alo[e]; a[e + 8] = ahi[e]; }
    // B 32x16 layout: lane<16 -> col N=lane, K {0..7,16..23}; lane>=16 -> K {8..15,24..31}
    v8h blo = *(const v8h*)(Br + k + hi * 8);
    v8h bhi = *(const v8h*)(Br + k + hi * 8 + 16);
    v16h bb;
#pragma unroll
    for (int e = 0; e < 8; ++e) { bb[e] = blo[e]; bb[e + 8] = bhi[e]; }
    acc = __builtin_amdgcn_wmma_f32_16x16x32_f16(false, a, false, bb,
                                                 (short)0, acc, false, false);
  }

  int gi = ti * 16, gj = tj * 16;
  size_t base = (size_t)b * Npad * Npad;
  float colp = 0.f;
#pragma unroll
  for (int r = 0; r < 8; ++r) {
    int m = r + hi * 8;                     // C/D: VGPR r, lane half selects M
    float s = __expf(10.0f * acc[r]);       // sim = exp(gamma * dot)
    S[base + (size_t)(gi + m) * Npad + (gj + laneM)] = s;
    colp += s;
    float rs = sum16(s);                    // sum over N within this tile
    if (laneM == 0) atomicAdd(&rowsum[b * Npad + gi + m], rs);
  }
  colp += __shfl_xor(colp, 16, 32);         // sum over all 16 M of the tile
  if (hi == 0) atomicAdd(&colsum[b * Npad + gj + laneM], colp);
}

// ------------- fuse cost: C = 1 - s / (colsum + rowsum - s), in place, b128 -------------
__global__ __launch_bounds__(256) void cost_fuse(float* __restrict__ S,
                                                 const float* __restrict__ rowsum,
                                                 const float* __restrict__ colsum)
{
  v4f* Sv = (v4f*)S;
  size_t total4 = (size_t)Bn * Npad * Npad / 4;
  for (size_t idx4 = (size_t)blockIdx.x * 256 + threadIdx.x; idx4 < total4;
       idx4 += (size_t)gridDim.x * 256) {
    size_t idx = idx4 * 4;
    int j0 = (int)(idx % Npad);            // multiple of 4
    size_t bi = idx / Npad;
    int b = (int)(bi / Npad); int i = (int)(bi % Npad);
    float rs = rowsum[b * Npad + i];
    v4f cs = *(const v4f*)(colsum + b * Npad + j0);
    v4f sv = Sv[idx4];
#pragma unroll
    for (int e = 0; e < 4; ++e) sv[e] = 1.0f - sv[e] / (cs[e] + rs - sv[e]);
    Sv[idx4] = sv;
  }
}

// ------------- hinge GEMM: sum relu(0.1 - x3@x4^T), WMMA + fused reduce -------------
__global__ __launch_bounds__(256) void hinge_gemm(const _Float16* __restrict__ x3h,
                                                  const _Float16* __restrict__ x4h,
                                                  float* __restrict__ out0)
{
  const int TPD = Nh / 16;  // 24
  int wid  = (blockIdx.x * blockDim.x + threadIdx.x) >> 5;
  int lane = threadIdx.x & 31;
  int b    = wid / (TPD * TPD);
  int rem  = wid % (TPD * TPD);
  int ti = rem / TPD, tj = rem % TPD;
  int laneM = lane & 15, hi = lane >> 4;

  const _Float16* Ar = x3h + ((size_t)b * Nh + ti * 16 + laneM) * Dd;
  const _Float16* Br = x4h + ((size_t)b * Nh + tj * 16 + laneM) * Dd;

  v8f acc = {};
#pragma unroll
  for (int k = 0; k < Dd; k += 32) {
    v8h alo = *(const v8h*)(Ar + k + hi * 8);
    v8h ahi = *(const v8h*)(Ar + k + hi * 8 + 16);
    v16h a;
#pragma unroll
    for (int e = 0; e < 8; ++e) { a[e] = alo[e]; a[e + 8] = ahi[e]; }
    v8h blo = *(const v8h*)(Br + k + hi * 8);
    v8h bhi = *(const v8h*)(Br + k + hi * 8 + 16);
    v16h bb;
#pragma unroll
    for (int e = 0; e < 8; ++e) { bb[e] = blo[e]; bb[e + 8] = bhi[e]; }
    acc = __builtin_amdgcn_wmma_f32_16x16x32_f16(false, a, false, bb,
                                                 (short)0, acc, false, false);
  }
  float p = 0.f;
#pragma unroll
  for (int r = 0; r < 8; ++r) p += fmaxf(0.1f - acc[r], 0.f);
  p = sum16(p);
  p += __shfl_xor(p, 16, 32);
  if (lane == 0) atomicAdd(out0, p);
}

// ------------- row softmin: out[b,i] = -eps*LSE_j( h(j) - C[b,i,j]/eps ) -------------
// h(j) = logmask(j) + g[b,j]*gscale (g may be null). One wave per row, float4 lanes.
__global__ __launch_bounds__(256) void softmin_rows(const float* __restrict__ C,
                                                    const float* __restrict__ g,
                                                    float gscale, float eps,
                                                    float* __restrict__ out)
{
  int wid  = (blockIdx.x * 256 + threadIdx.x) >> 5;
  int lane = threadIdx.x & 31;
  int b = wid / Npad, i = wid % Npad;
  const v4f* rowv = (const v4f*)(C + ((size_t)b * Npad + i) * Npad);  // 224 float4/row
  const v4f* gv   = g ? (const v4f*)(g + b * Npad) : nullptr;
  float inv_eps = 1.0f / eps;
  float m = -3.0e38f, s = 0.f;
#pragma unroll
  for (int c = lane; c < Npad / 4; c += 32) {     // 7 iterations
    v4f q = rowv[c];
    __builtin_prefetch(rowv + c + 32, 0, 0);
    v4f gq = {};
    if (gv) gq = gv[c];
    int j0 = c * 4;
#pragma unroll
    for (int e = 0; e < 4; ++e) {
      float h = (j0 + e < Nreal ? 0.f : NEGW) + gq[e] * gscale;
      lse_acc(m, s, h - q[e] * inv_eps);
    }
  }
#pragma unroll
  for (int d = 16; d >= 1; d >>= 1) {
    float om = __shfl_xor(m, d, 32);
    float os = __shfl_xor(s, d, 32);
    float nm = fmaxf(m, om);
    s = s * __expf(m - nm) + os * __expf(om - nm);
    m = nm;
  }
  if (lane == 0) out[b * Npad + i] = -eps * (m + __logf(s));
}

// ------------- col softmin: out[b,j] = -eps*LSE_i( h(i) - C[b,i,j]/eps ) -------------
// 16x16 threads; each thread owns 4 consecutive columns (float4 loads, coalesced),
// 16 row-strips, LDS combine. 64 columns per block.
__global__ __launch_bounds__(256) void softmin_cols(const float* __restrict__ C,
                                                    const float* __restrict__ f,
                                                    float fscale, float eps,
                                                    float* __restrict__ out)
{
  __shared__ float sm[16][64], ss[16][64];
  int tx = threadIdx.x & 15;     // column group
  int ty = threadIdx.x >> 4;     // row strip
  int b  = blockIdx.x / (Npad / 64);
  int jg = blockIdx.x % (Npad / 64);
  int j0 = jg * 64 + tx * 4;
  const float* base = C + (size_t)b * Npad * Npad;
  float inv_eps = 1.0f / eps;
  float m[4], s[4];
#pragma unroll
  for (int e = 0; e < 4; ++e) { m[e] = -3.0e38f; s[e] = 0.f; }
  for (int i = ty; i < Npad; i += 16) {           // 56 iterations
    float h = (i < Nreal ? 0.f : NEGW);
    if (f) h += f[b * Npad + i] * fscale;
    v4f q = *(const v4f*)(base + (size_t)i * Npad + j0);
#pragma unroll
    for (int e = 0; e < 4; ++e) lse_acc(m[e], s[e], h - q[e] * inv_eps);
  }
#pragma unroll
  for (int e = 0; e < 4; ++e) { sm[ty][tx * 4 + e] = m[e]; ss[ty][tx * 4 + e] = s[e]; }
  __syncthreads();
  if (threadIdx.x < 64) {
    int jj = threadIdx.x;
    float M = sm[0][jj], S = ss[0][jj];
#pragma unroll
    for (int t = 1; t < 16; ++t) {
      float om = sm[t][jj], os = ss[t][jj];
      float nm = fmaxf(M, om);
      S = S * __expf(M - nm) + os * __expf(om - nm);
      M = nm;
    }
    out[b * Npad + jg * 64 + jj] = -eps * (M + __logf(S));
  }
}

// ------------- dual averaging: f = 0.5*(f+ft), g = 0.5*(g+gt) -------------
__global__ void avg_update(float* __restrict__ fba, const float* __restrict__ tf,
                           float* __restrict__ gab, const float* __restrict__ tg, int n)
{
  int i = blockIdx.x * blockDim.x + threadIdx.x;
  if (i < n) {
    fba[i] = 0.5f * (fba[i] + tf[i]);
    gab[i] = 0.5f * (gab[i] + tg[i]);
  }
}

// ------------- final: scon = mean_b sum_{i<512}(f_new + g_new) -------------
__global__ __launch_bounds__(256) void final_cost(const float* __restrict__ fnew,
                                                  const float* __restrict__ gnew,
                                                  float* __restrict__ out1)
{
  __shared__ float red[256];
  float acc = 0.f;
  const int total = Bn * Nreal;
  for (int t = threadIdx.x; t < total; t += 256) {
    int b = t / Nreal, i = t % Nreal;
    acc += fnew[b * Npad + i] + gnew[b * Npad + i];
  }
  red[threadIdx.x] = acc;
  __syncthreads();
  for (int d = 128; d > 0; d >>= 1) {
    if (threadIdx.x < d) red[threadIdx.x] += red[threadIdx.x + d];
    __syncthreads();
  }
  if (threadIdx.x == 0) out1[0] = red[0] / (float)Bn;
}

extern "C" void kernel_launch(void* const* d_in, const int* in_sizes, int n_in,
                              void* d_out, int out_size, void* d_ws, size_t ws_size,
                              hipStream_t stream)
{
  (void)in_sizes; (void)n_in; (void)out_size; (void)ws_size;
  const float* x1 = (const float*)d_in[0];
  const float* x2 = (const float*)d_in[1];
  const float* x3 = (const float*)d_in[2];
  const float* x4 = (const float*)d_in[3];
  float* out = (float*)d_out;

  char* ws = (char*)d_ws;
  size_t off = 0;
  auto carve = [&](size_t bytes) -> char* {
    char* p = ws + off;
    off = (off + bytes + 255) & ~(size_t)255;
    return p;
  };
  _Float16* f1h = (_Float16*)carve((size_t)Bn * Npad * Dd * 2);
  _Float16* f2h = (_Float16*)carve((size_t)Bn * Npad * Dd * 2);
  _Float16* x3h = (_Float16*)carve((size_t)Bn * Nh * Dd * 2);
  _Float16* x4h = (_Float16*)carve((size_t)Bn * Nh * Dd * 2);
  float* S      = (float*)carve((size_t)Bn * Npad * Npad * 4);   // sim -> cost, 51 MB (L2-resident)
  float* rowsum = (float*)carve((size_t)Bn * Npad * 4);
  float* colsum = (float*)carve((size_t)Bn * Npad * 4);
  float* fba    = (float*)carve((size_t)Bn * Npad * 4);
  float* gab    = (float*)carve((size_t)Bn * Npad * 4);
  float* tf     = (float*)carve((size_t)Bn * Npad * 4);
  float* tg     = (float*)carve((size_t)Bn * Npad * 4);
  float* fnew   = (float*)carve((size_t)Bn * Npad * 4);
  float* gnew   = (float*)carve((size_t)Bn * Npad * 4);

  init_zero<<<(Bn * Npad + 255) / 256, 256, 0, stream>>>(rowsum, colsum, out);
  prep_norm<<<(2 * Bn * Npad) / 8, 256, 0, stream>>>(x1, x2, x3, x4, f1h, f2h, x3h, x4h);
  sim_gemm<<<(Bn * 56 * 56) / 8, 256, 0, stream>>>(f1h, f2h, S, rowsum, colsum);
  cost_fuse<<<2048, 256, 0, stream>>>(S, rowsum, colsum);
  hinge_gemm<<<(Bn * 24 * 24) / 8, 256, 0, stream>>>(x3h, x4h, &out[0]);

  // geomloss epsilon schedule for p=2, diameter=3, blur=0.05, scaling=0.5
  const float EPSL[8] = {9.0f, 9.0f, 2.25f, 0.5625f,
                         0.140625f, 0.03515625f, 0.0087890625f, 0.0025f};
  const int rsGrid = (Bn * Npad) / 8;   // wave per row
  const int csGrid = Bn * (Npad / 64);  // 64 cols per block

  // initial duals at eps = diameter^2
  softmin_rows<<<rsGrid, 256, 0, stream>>>(S, nullptr, 0.f, 9.0f, fba);  // f_ba
  softmin_cols<<<csGrid, 256, 0, stream>>>(S, nullptr, 0.f, 9.0f, gab);  // g_ab

  for (int it = 0; it < 8; ++it) {
    float e = EPSL[it], ie = 1.0f / e;
    softmin_rows<<<rsGrid, 256, 0, stream>>>(S, gab, ie, e, tf);   // ft
    softmin_cols<<<csGrid, 256, 0, stream>>>(S, fba, ie, e, tg);   // gt
    avg_update<<<(Bn * Npad + 255) / 256, 256, 0, stream>>>(fba, tf, gab, tg, Bn * Npad);
  }

  // last extrapolation at eps = blur^2
  const float el = 0.0025f, iel = 1.0f / el;
  softmin_rows<<<rsGrid, 256, 0, stream>>>(S, gab, iel, el, fnew);
  softmin_cols<<<csGrid, 256, 0, stream>>>(S, fba, iel, el, gnew);

  final_cost<<<1, 256, 0, stream>>>(fnew, gnew, &out[1]);
}
